// HMM_DPM_61787399520276
// MI455X (gfx1250) — matrix operand order repaired and checked by reading, not compiled, not run on hardware
//
#include <hip/hip_runtime.h>
#include <hip/hip_bf16.h>
#include <math.h>

// ---------------------------------------------------------------------------
// Problem constants (from reference)
// ---------------------------------------------------------------------------
#define BB      16
#define LL      256
#define DD      64
#define HH      256
#define KK      3
#define SS      8
#define NN      (BB * LL)       // 4096
#define NSTEPS  50
#define NB      (NN / 16)       // 256 row-tiles per s

typedef __attribute__((ext_vector_type(16))) __bf16 v16bf;
typedef __attribute__((ext_vector_type(8)))  __bf16 v8bf;
typedef __attribute__((ext_vector_type(8)))  float  v8f;

__device__ __forceinline__ float silu(float x) { return x / (1.f + __expf(-x)); }

// ---------------------------------------------------------------------------
// A-fragment load (16-bit A 16x32 layout, CDNA5 ISA 7.12.2):
//   lanes 0-15  hold K = [k0+0..8) and [k0+16..24)
//   lanes 16-31 hold K = [k0+8..16) and [k0+24..32)
// rowp = LDS row base (row = lane&15), half = lane>>4
// ---------------------------------------------------------------------------
__device__ __forceinline__ v16bf load_a_frag(const __bf16* rowp, int k0, int half) {
  v8bf lo = *(const v8bf*)(rowp + k0 + half * 8);
  v8bf hi = *(const v8bf*)(rowp + k0 + 16 + half * 8);
  v16bf a;
#pragma unroll
  for (int i = 0; i < 8; ++i) { a[i] = lo[i]; a[i + 8] = hi[i]; }
  return a;
}

// ---------------------------------------------------------------------------
// One wave computes nTiles 16x16 output tiles of D = A(16xK) * B(Kx16tiles).
// A is row-major bf16 in LDS; Bt is the PRE-TRANSPOSED weight [outCol][K] in
// global memory, so each lane's B fragment (col = col0 + lane&15,
// K = [k0+(lane>>4)*8..+8) and [k0+16+(lane>>4)*8..+8)) is two 16-byte loads.
// Epilogue: epi(col, rowBase, acc) with rows rowBase..rowBase+7.
// ---------------------------------------------------------------------------
template <class Epi>
__device__ __forceinline__ void wave_gemm16(const __bf16* sA, int Kdim,
                                            const __bf16* Bt, int colBase,
                                            int nTiles, int lane, Epi epi) {
  const int half = lane >> 4;
  const int lidx = lane & 15;
  const __bf16* rowp = sA + lidx * Kdim;
  for (int t = 0; t < nTiles; ++t) {
    const int col0 = colBase + (t << 4);
    const __bf16* bcol = Bt + (size_t)(col0 + lidx) * Kdim;
    // hint: pull next tile's column stream toward GL2/L0
    __builtin_prefetch(bcol + 16 * Kdim, 0, 1);
    v8f acc = {};
    for (int k0 = 0; k0 < Kdim; k0 += 32) {
      v16bf a = load_a_frag(rowp, k0, half);
      v16bf b = load_a_frag(bcol, k0, half);
      acc = __builtin_amdgcn_wmma_f32_16x16x32_bf16(
          /*neg_a=*/false, a, /*neg_b=*/false, b,
          /*c_mod=*/(short)0, acc, /*reuse_a=*/false, /*reuse_b=*/false);
    }
    epi(col0 + lidx, half * 8, acc);
  }
}

// ---------------------------------------------------------------------------
// Prep kernel 1: f32 weight [R][C] -> bf16 transposed [C][R]
// ---------------------------------------------------------------------------
__global__ __launch_bounds__(256) void wcvt(const float* __restrict__ W,
                                            __bf16* __restrict__ Wt, int R, int C) {
  int i = blockIdx.x * blockDim.x + threadIdx.x;   // linear over output [C][R]
  if (i >= R * C) return;
  int c = i / R, r = i - c * R;
  Wt[i] = (__bf16)W[r * C + c];
}

// ---------------------------------------------------------------------------
// Prep kernel 2: abar cumprod + state_emb @ adaln{1,2}_proj (tiny GEMMs, f32)
// ---------------------------------------------------------------------------
__global__ __launch_bounds__(256) void prep_small(
    const float* __restrict__ state_emb,
    const float* __restrict__ a1w, const float* __restrict__ a1b,
    const float* __restrict__ a2w, const float* __restrict__ a2b,
    float* __restrict__ abar, float* __restrict__ se1, float* __restrict__ se2) {
  int tid = threadIdx.x;
  if (tid == 0) {
    float p = 1.f;
    for (int t = 0; t < NSTEPS; ++t) {
      float beta = 1e-4f + (2e-2f - 1e-4f) * (float)t / (float)(NSTEPS - 1);
      p *= (1.f - beta);
      abar[t] = p;
    }
  }
  for (int j = tid; j < KK * 4 * HH; j += blockDim.x) {   // 3 x 1024
    int k = j >> 10, c = j & 1023;
    float s = a1b[c];
    for (int r = 0; r < HH; ++r) s += state_emb[k * HH + r] * a1w[r * 4 * HH + c];
    se1[j] = s;
  }
  for (int j = tid; j < KK * 2 * HH; j += blockDim.x) {   // 3 x 512
    int k = j >> 9, c = j & 511;
    float s = a2b[c];
    for (int r = 0; r < HH; ++r) s += state_emb[k * HH + r] * a2w[r * 2 * HH + c];
    se2[j] = s;
  }
}

// ---------------------------------------------------------------------------
// Main fused kernel: one block = 16 rows of (s, n) space, 4 wave32s.
// Full trunk + per-state head fused in LDS; writes per-(s,n,k) SSE partials.
// ---------------------------------------------------------------------------
// dynamic LDS layout (bytes)
#define O_TEMBR 0u          // bf16 16x256  ( 8192)
#define O_TEMBB 8192u       // bf16 16x256  ( 8192)
#define O_XT    16384u      // bf16 16x64   ( 2048)
#define O_HS    18432u      // bf16 16x512  (16384)
#define O_HN    34816u      // f32  16x512  (32768)
#define O_W1    67584u      // f32  16x1024 (65536)
#define O_W2    133120u     // f32  16x512  (32768)
#define O_G     165888u     // f32  16x256  (16384)
#define O_H3    182272u     // bf16 16x256  ( 8192)
#define O_PRED  190464u     // f32  16x64   ( 4096)
#define O_MEAN  194560u     // f32  16      (   64)
#define O_VAR   194624u     // f32  16      (   64)
#define O_AB    194688u     // f32  16      (   64)
#define O_T     194752u     // f32  16      (   64)
#define SMEM_BYTES 194816u

__global__ __launch_bounds__(128) void fused_dpm(
    const float* __restrict__ y, const int* __restrict__ t_samples,
    const float* __restrict__ noise,
    const float* __restrict__ time_b, const float* __restrict__ inproj_b,
    const float* __restrict__ lin1_b,
    const float* __restrict__ n1w, const float* __restrict__ n1b,
    const float* __restrict__ a1b,
    const float* __restrict__ lin2_b,
    const float* __restrict__ n2w, const float* __restrict__ n2b,
    const float* __restrict__ a2b,
    const float* __restrict__ out_b,
    const float* __restrict__ abar,
    const float* __restrict__ se1, const float* __restrict__ se2,
    const __bf16* __restrict__ time_wt, const __bf16* __restrict__ inproj_wt,
    const __bf16* __restrict__ lin1_wt, const __bf16* __restrict__ adaln1_wt,
    const __bf16* __restrict__ lin2_wt, const __bf16* __restrict__ adaln2_wt,
    const __bf16* __restrict__ out_wt,
    float* __restrict__ ssePart) {
  const int sIdx = blockIdx.x >> 8;        // 0..S-1
  const int nb   = blockIdx.x & (NB - 1);  // 0..255
  const int n0   = nb * 16;
  const int row0 = sIdx * NN + n0;         // row in (S*N)

  const int tid  = threadIdx.x;
  const int lane = tid & 31;
  const int wv   = tid >> 5;

  extern __shared__ __align__(64) char smem[];
  __bf16* aTembR = (__bf16*)(smem + O_TEMBR);
  __bf16* aTembB = (__bf16*)(smem + O_TEMBB);
  __bf16* aXt    = (__bf16*)(smem + O_XT);
  __bf16* aHs    = (__bf16*)(smem + O_HS);
  float*  aHn    = (float*)(smem + O_HN);
  float*  aW1    = (float*)(smem + O_W1);
  float*  aW2    = (float*)(smem + O_W2);
  float*  aG     = (float*)(smem + O_G);
  __bf16* aH3    = (__bf16*)(smem + O_H3);
  float*  aPred  = (float*)(smem + O_PRED);
  float*  aMean  = (float*)(smem + O_MEAN);
  float*  aVar   = (float*)(smem + O_VAR);
  float*  aAb    = (float*)(smem + O_AB);
  float*  aT     = (float*)(smem + O_T);

  // ---- Step A: per-row t, abar; x_t (bf16); sinusoidal temb (bf16) --------
  if (tid < 16) {
    int t = t_samples[row0 + tid];
    aT[tid]  = (float)t;
    aAb[tid] = abar[t];
  }
  __syncthreads();
  for (int i = tid; i < 16 * DD; i += 128) {
    int r = i >> 6, d = i & 63;
    float ab = aAb[r];
    float xv = y[(size_t)(n0 + r) * DD + d];
    float nv = noise[(size_t)(row0 + r) * DD + d];
    aXt[i] = (__bf16)(sqrtf(ab) * xv + sqrtf(1.f - ab) * nv);
  }
  const float lf = -9.210340371976184f / 127.0f;   // -ln(10000)/(half-1)
  for (int i = tid; i < 16 * 128; i += 128) {
    int r = i >> 7, j = i & 127;
    float ang = aT[r] * __expf((float)j * lf);
    aTembR[r * HH + j]       = (__bf16)__sinf(ang);
    aTembR[r * HH + 128 + j] = (__bf16)__cosf(ang);
  }
  __syncthreads();

  // ---- Step B: temb = silu(tembRaw @ time_w + time_b)  (16x256x256) ------
  wave_gemm16(aTembR, HH, time_wt, wv * 64, 4, lane,
              [&](int col, int rb, v8f acc) {
                float bb = time_b[col];
#pragma unroll
                for (int v = 0; v < 8; ++v)
                  aTembB[(rb + v) * HH + col] = (__bf16)silu(acc[v] + bb);
              });
  __syncthreads();

  // ---- Step C: h1 = silu(x_t @ inproj_w + b)  (16x64x512) ----------------
  wave_gemm16(aXt, DD, inproj_wt, wv * 128, 8, lane,
              [&](int col, int rb, v8f acc) {
                float bb = inproj_b[col];
#pragma unroll
                for (int v = 0; v < 8; ++v)
                  aHs[(rb + v) * 512 + col] = (__bf16)silu(acc[v] + bb);
              });
  __syncthreads();

  // ---- Step D: h2 = h1s @ lin1_w + b  (16x512x512) -> f32 ----------------
  wave_gemm16(aHs, 512, lin1_wt, wv * 128, 8, lane,
              [&](int col, int rb, v8f acc) {
                float bb = lin1_b[col];
#pragma unroll
                for (int v = 0; v < 8; ++v)
                  aHn[(rb + v) * 512 + col] = acc[v] + bb;
              });
  __syncthreads();

  // ---- Step E: layernorm over 2H=512, in place ---------------------------
  if (tid < 16) { aMean[tid] = 0.f; aVar[tid] = 0.f; }
  __syncthreads();
  {
    int r = tid >> 3, c0 = (tid & 7) * 64;
    float s = 0.f, ss = 0.f;
    for (int c = c0; c < c0 + 64; ++c) { float v = aHn[r * 512 + c]; s += v; ss += v * v; }
    atomicAdd(&aMean[r], s);
    atomicAdd(&aVar[r], ss);
  }
  __syncthreads();
  {
    int r = tid >> 3, c0 = (tid & 7) * 64;
    float m = aMean[r] * (1.f / 512.f);
    float var = aVar[r] * (1.f / 512.f) - m * m;
    float rstd = rsqrtf(var + 1e-5f);
    for (int c = c0; c < c0 + 64; ++c)
      aHn[r * 512 + c] = (aHn[r * 512 + c] - m) * rstd * n1w[c] + n1b[c];
  }
  __syncthreads();

  // ---- Steps F,G: AdaLN projections of temb (state part precomputed) -----
  wave_gemm16(aTembB, HH, adaln1_wt, wv * 256, 16, lane,   // 16x256x1024
              [&](int col, int rb, v8f acc) {
                float bb = a1b[col];
#pragma unroll
                for (int v = 0; v < 8; ++v) aW1[(rb + v) * 1024 + col] = acc[v] + bb;
              });
  wave_gemm16(aTembB, HH, adaln2_wt, wv * 128, 8, lane,    // 16x256x512
              [&](int col, int rb, v8f acc) {
                float bb = a2b[col];
#pragma unroll
                for (int v = 0; v < 8; ++v) aW2[(rb + v) * 512 + col] = acc[v] + bb;
              });
  __syncthreads();

  // ---- Step H: per-state head -------------------------------------------
  for (int k = 0; k < KK; ++k) {
    const float* e1 = se1 + k * 1024;
    const float* e2 = se2 + k * 512;

    // modulate hn with AdaLN1, silu -> bf16
    for (int i = tid; i < 16 * 512; i += 128) {
      int r = i >> 9, c = i & 511;
      float sc = aW1[r * 1024 + c] + e1[c];
      float sh = aW1[r * 1024 + 512 + c] + e1[512 + c];
      aHs[i] = (__bf16)silu(aHn[i] * (1.f + sc) + sh);
    }
    __syncthreads();

    // lin2: 16x512x256 -> f32
    wave_gemm16(aHs, 512, lin2_wt, wv * 64, 4, lane,
                [&](int col, int rb, v8f acc) {
                  float bb = lin2_b[col];
#pragma unroll
                  for (int v = 0; v < 8; ++v) aG[(rb + v) * HH + col] = acc[v] + bb;
                });
    __syncthreads();

    // layernorm over H=256 + AdaLN2 modulate + silu -> bf16
    if (tid < 16) { aMean[tid] = 0.f; aVar[tid] = 0.f; }
    __syncthreads();
    {
      int r = tid >> 3, c0 = (tid & 7) * 32;
      float s = 0.f, ss = 0.f;
      for (int c = c0; c < c0 + 32; ++c) { float v = aG[r * HH + c]; s += v; ss += v * v; }
      atomicAdd(&aMean[r], s);
      atomicAdd(&aVar[r], ss);
    }
    __syncthreads();
    {
      int r = tid >> 3, c0 = (tid & 7) * 32;
      float m = aMean[r] * (1.f / 256.f);
      float var = aVar[r] * (1.f / 256.f) - m * m;
      float rstd = rsqrtf(var + 1e-5f);
      for (int c = c0; c < c0 + 32; ++c) {
        float hn2 = (aG[r * HH + c] - m) * rstd * n2w[c] + n2b[c];
        float sc = aW2[r * 512 + c] + e2[c];
        float sh = aW2[r * 512 + 256 + c] + e2[256 + c];
        aH3[r * HH + c] = (__bf16)silu(hn2 * (1.f + sc) + sh);
      }
    }
    __syncthreads();

    // out: 16x256x64 -> pred f32
    wave_gemm16(aH3, HH, out_wt, wv * 16, 1, lane,
                [&](int col, int rb, v8f acc) {
                  float bb = out_b[col];
#pragma unroll
                  for (int v = 0; v < 8; ++v) aPred[(rb + v) * DD + col] = acc[v] + bb;
                });
    __syncthreads();

    // sse per row (deterministic partial write, no atomics)
    if (tid < 16) {
      int r = tid;
      const float* np = noise + (size_t)(row0 + r) * DD;
      float s = 0.f;
      for (int d = 0; d < DD; ++d) {
        float df = aPred[r * DD + d] - np[d];
        s += df * df;
      }
      ssePart[(size_t)(row0 + r) * KK + k] = s;
    }
    __syncthreads();
  }
}

// ---------------------------------------------------------------------------
// HMM forward recursion (log-space alpha), one block per batch element.
// ---------------------------------------------------------------------------
__global__ __launch_bounds__(32) void hmm_forward(
    const float* __restrict__ ssePart, const float* __restrict__ trans_logits,
    const float* __restrict__ start_logits, float* __restrict__ out) {
  if (threadIdx.x != 0) return;
  int b = blockIdx.x;
  float lt[KK][KK], ls[KK];
  for (int i = 0; i < KK; ++i) {
    float m = fmaxf(fmaxf(trans_logits[i * 3], trans_logits[i * 3 + 1]), trans_logits[i * 3 + 2]);
    float z = 0.f;
    for (int j = 0; j < KK; ++j) z += __expf(trans_logits[i * 3 + j] - m);
    float l = __logf(z) + m;
    for (int j = 0; j < KK; ++j) lt[i][j] = trans_logits[i * 3 + j] - l;
  }
  {
    float m = fmaxf(fmaxf(start_logits[0], start_logits[1]), start_logits[2]);
    float z = 0.f;
    for (int j = 0; j < KK; ++j) z += __expf(start_logits[j] - m);
    float l = __logf(z) + m;
    for (int j = 0; j < KK; ++j) ls[j] = start_logits[j] - l;
  }
  float a[KK];
  for (int l = 0; l < LL; ++l) {
    int n = b * LL + l;
    float em[KK];
    for (int kk = 0; kk < KK; ++kk) {
      float s = 0.f;
      for (int ss = 0; ss < SS; ++ss) s += ssePart[((size_t)ss * NN + n) * KK + kk];
      em[kk] = -0.5f * s * (1.f / (float)SS);
    }
    if (l == 0) {
      for (int kk = 0; kk < KK; ++kk) a[kk] = ls[kk] + em[kk];
    } else {
      float na[KK];
      for (int kk = 0; kk < KK; ++kk) {
        float v0 = a[0] + lt[0][kk], v1 = a[1] + lt[1][kk], v2 = a[2] + lt[2][kk];
        float m = fmaxf(fmaxf(v0, v1), v2);
        na[kk] = em[kk] + __logf(__expf(v0 - m) + __expf(v1 - m) + __expf(v2 - m)) + m;
      }
      for (int kk = 0; kk < KK; ++kk) a[kk] = na[kk];
    }
    for (int kk = 0; kk < KK; ++kk) out[(size_t)n * KK + kk] = a[kk];
  }
}

// ---------------------------------------------------------------------------
// Workspace layout (byte offsets, all 256B aligned)
// ---------------------------------------------------------------------------
static constexpr size_t OFF_ABAR  = 0;        //   64 f32
static constexpr size_t OFF_SE1   = 256;      // 3072 f32
static constexpr size_t OFF_SE2   = 12544;    // 1536 f32
static constexpr size_t OFF_SSE   = 18688;    // S*N*K f32 = 98304
static constexpr size_t OFF_TIMEW = 411904;   //  65536 bf16
static constexpr size_t OFF_INPW  = 542976;   //  32768 bf16
static constexpr size_t OFF_L1W   = 608512;   // 262144 bf16
static constexpr size_t OFF_A1W   = 1132800;  // 262144 bf16
static constexpr size_t OFF_L2W   = 1657088;  // 131072 bf16
static constexpr size_t OFF_A2W   = 1919232;  // 131072 bf16
static constexpr size_t OFF_OUTW  = 2181376;  //  16384 bf16

extern "C" void kernel_launch(void* const* d_in, const int* in_sizes, int n_in,
                              void* d_out, int out_size, void* d_ws, size_t ws_size,
                              hipStream_t stream) {
  (void)in_sizes; (void)n_in; (void)out_size; (void)ws_size;
  const float* y            = (const float*)d_in[0];
  const int*   t_samples    = (const int*)d_in[1];
  const float* noise        = (const float*)d_in[2];
  const float* time_w       = (const float*)d_in[3];
  const float* time_b       = (const float*)d_in[4];
  const float* state_emb    = (const float*)d_in[5];
  const float* inproj_w     = (const float*)d_in[6];
  const float* inproj_b     = (const float*)d_in[7];
  const float* lin1_w       = (const float*)d_in[8];
  const float* lin1_b       = (const float*)d_in[9];
  const float* n1w          = (const float*)d_in[10];
  const float* n1b          = (const float*)d_in[11];
  const float* a1wF         = (const float*)d_in[12];
  const float* a1b          = (const float*)d_in[13];
  const float* lin2_w       = (const float*)d_in[14];
  const float* lin2_b       = (const float*)d_in[15];
  const float* n2w          = (const float*)d_in[16];
  const float* n2b          = (const float*)d_in[17];
  const float* a2wF         = (const float*)d_in[18];
  const float* a2b          = (const float*)d_in[19];
  const float* out_w        = (const float*)d_in[20];
  const float* out_b        = (const float*)d_in[21];
  const float* trans_logits = (const float*)d_in[22];
  const float* start_logits = (const float*)d_in[23];

  char* ws = (char*)d_ws;
  float*  abar    = (float*)(ws + OFF_ABAR);
  float*  se1     = (float*)(ws + OFF_SE1);
  float*  se2     = (float*)(ws + OFF_SE2);
  float*  ssePart = (float*)(ws + OFF_SSE);
  __bf16* time_wt   = (__bf16*)(ws + OFF_TIMEW);
  __bf16* inproj_wt = (__bf16*)(ws + OFF_INPW);
  __bf16* lin1_wt   = (__bf16*)(ws + OFF_L1W);
  __bf16* adaln1_wt = (__bf16*)(ws + OFF_A1W);
  __bf16* lin2_wt   = (__bf16*)(ws + OFF_L2W);
  __bf16* adaln2_wt = (__bf16*)(ws + OFF_A2W);
  __bf16* out_wt    = (__bf16*)(ws + OFF_OUTW);

  // 1) weight conversion + transpose to bf16 [C][R]
  auto cvt = [&](const float* W, __bf16* Wt, int R, int C) {
    int n = R * C;
    wcvt<<<(n + 255) / 256, 256, 0, stream>>>(W, Wt, R, C);
  };
  cvt(time_w,   time_wt,   HH,  HH);        // 256x256
  cvt(inproj_w, inproj_wt, DD,  2 * HH);    // 64x512
  cvt(lin1_w,   lin1_wt,   2 * HH, 2 * HH); // 512x512
  cvt(a1wF,     adaln1_wt, HH,  4 * HH);    // 256x1024
  cvt(lin2_w,   lin2_wt,   2 * HH, HH);     // 512x256
  cvt(a2wF,     adaln2_wt, HH,  2 * HH);    // 256x512
  cvt(out_w,    out_wt,    HH,  DD);        // 256x64

  // 2) abar + state-embedding projections
  prep_small<<<1, 256, 0, stream>>>(state_emb, a1wF, a1b, a2wF, a2b, abar, se1, se2);

  // 3) fused trunk + heads (WMMA)
  fused_dpm<<<dim3(SS * NB), dim3(128), SMEM_BYTES, stream>>>(
      y, t_samples, noise,
      time_b, inproj_b, lin1_b, n1w, n1b, a1b, lin2_b, n2w, n2b, a2b, out_b,
      abar, se1, se2,
      time_wt, inproj_wt, lin1_wt, adaln1_wt, lin2_wt, adaln2_wt, out_wt,
      ssePart);

  // 4) HMM forward recursion -> alpha (B,L,K) f32
  hmm_forward<<<dim3(BB), dim3(32), 0, stream>>>(
      ssePart, trans_logits, start_logits, (float*)d_out);
}
